// EmbeddingUpdater_48610439856734
// MI455X (gfx1250) — matrix operand order repaired and checked by reading, not compiled, not run on hardware
//
#include <hip/hip_runtime.h>
#include <hip/hip_fp16.h>

typedef __attribute__((ext_vector_type(16))) _Float16 v16h;
typedef __attribute__((ext_vector_type(8)))  _Float16 v8h;
typedef __attribute__((ext_vector_type(8)))  float    v8f;

#define HID   512
#define NB    20000
#define NN    100000
#define G3H   (3 * HID)

// ---------------------------------------------------------------------------
// WMMA helpers (CDNA5 gfx1250, wave32, 16x16x32 f16 -> f32 accum)
// ---------------------------------------------------------------------------
__device__ __forceinline__ v8f wmma_f16(v16h a, v16h b, v8f c) {
  // 8 args: (neg_a, A, neg_b, B, c_mod, C, reuse_a, reuse_b)
  return __builtin_amdgcn_wmma_f32_16x16x32_f16(false, a, false, b, (short)0, c,
                                                false, false);
}

// A fragment: 16(M) x 32(K) f16, row-major source with leading dim `ld`.
// Lanes 0-15 : row m0+lane, halves = K {k0..k0+7} then {k0+16..k0+23}
// Lanes 16-31: row m0+lane-16, halves = K {k0+8..k0+15} then {k0+24..k0+31}
__device__ __forceinline__ v16h load_a_frag(const _Float16* __restrict__ base,
                                            int ld, int m0, int k0, int lane) {
  int r = m0 + (lane & 15);
  int ks = k0 + ((lane >> 4) << 3);  // +0 or +8
  const _Float16* p = base + (size_t)r * ld + ks;
  v8h lo = *(const v8h*)(p);
  v8h hi = *(const v8h*)(p + 16);
  v16h a;
#pragma unroll
  for (int i = 0; i < 8; ++i) { a[i] = lo[i]; a[i + 8] = hi[i]; }
  return a;
}

// B fragment: 32(K) x 16(N). B[k][n] = W[n][k] with W row-major [rows][ld].
// Lanes 0-15 : column n0+lane, K {k0..k0+7} then {k0+16..k0+23}
// Lanes 16-31: column n0+lane-16, K {k0+8..k0+15} then {k0+24..k0+31}
__device__ __forceinline__ v16h load_b_frag(const _Float16* __restrict__ W,
                                            int ld, int n0, int k0, int lane) {
  int wrow = n0 + (lane & 15);
  int ks = k0 + ((lane >> 4) << 3);  // +0 or +8
  const _Float16* p = W + (size_t)wrow * ld + ks;
  v8h lo = *(const v8h*)(p);
  v8h hi = *(const v8h*)(p + 16);
  v16h b;
#pragma unroll
  for (int i = 0; i < 8; ++i) { b[i] = lo[i]; b[i + 8] = hi[i]; }
  return b;
}

__device__ __forceinline__ float sigf(float x) {
  return 1.0f / (1.0f + __expf(-x));
}

// GRU epilogue for one 16x16 C/D tile half (this lane owns column n,
// rows mbase..mbase+7 in accumulator elements 0..7).
__device__ __forceinline__ void gru_epilogue(
    v8f air, v8f aiz, v8f ain, v8f ahr, v8f ahz, v8f ahn, int mbase, int n,
    float bi_r, float bi_z, float bi_n, float bh_r, float bh_z, float bh_n,
    const int* __restrict__ nids, const float* __restrict__ dyn, int layer,
    float* __restrict__ out, _Float16* __restrict__ Hnext) {
#pragma unroll
  for (int v = 0; v < 8; ++v) {
    int row = mbase + v;
    int nid = nids[row];
    size_t oidx = (size_t)nid * (2 * HID) + (size_t)layer * HID + n;
    float hprev = dyn[oidx];
    float r = sigf(air[v] + bi_r + ahr[v] + bh_r);
    float z = sigf(aiz[v] + bi_z + ahz[v] + bh_z);
    float ng = tanhf(ain[v] + bi_n + r * (ahn[v] + bh_n));
    float hn = (1.0f - z) * ng + z * hprev;
    out[oidx] = hn;
    if (Hnext) Hnext[(size_t)row * HID + n] = (_Float16)hn;
  }
}

// ---------------------------------------------------------------------------
// Fused GRU layer: h_new = GRU(X, Hp) ; scatter f32 result into out[nid][layer]
// Block = 256 threads = 8 waves. Block tile: 32 rows x 128 cols.
// Wave tile: 32(M) x 16(N) -> 12 v8f accumulators (3 gates x {gi,gh} x 2).
// ---------------------------------------------------------------------------
__global__ __launch_bounds__(256) void gru_layer_kernel(
    const _Float16* __restrict__ X, int Kx,          // [NB, Kx] f16
    const _Float16* __restrict__ Hp,                 // [NB, HID] f16
    const _Float16* __restrict__ Wih,                // [3H, Kx] f16
    const _Float16* __restrict__ Whh,                // [3H, HID] f16
    const float* __restrict__ bih, const float* __restrict__ bhh,
    const int* __restrict__ nids, const float* __restrict__ dyn, int layer,
    float* __restrict__ out, _Float16* __restrict__ Hnext) {
  const int lane = threadIdx.x & 31;
  const int wave = threadIdx.x >> 5;
  const int colBlk = blockIdx.x & 3;       // 4 col blocks of 128
  const int rowBlk = blockIdx.x >> 2;      // 625 row blocks of 32
  const int n0 = colBlk * 128 + wave * 16;
  const int m0 = rowBlk * 32;

  v8f i0r = {}, i0z = {}, i0n = {}, i1r = {}, i1z = {}, i1n = {};
  v8f h0r = {}, h0z = {}, h0n = {}, h1r = {}, h1z = {}, h1n = {};

  // gi = X @ Wih^T   (gate columns n0 / n0+H / n0+2H)
#pragma unroll 4
  for (int k = 0; k < Kx; k += 32) {
    v16h a0 = load_a_frag(X, Kx, m0, k, lane);
    v16h a1 = load_a_frag(X, Kx, m0 + 16, k, lane);
    v16h br = load_b_frag(Wih, Kx, n0, k, lane);
    v16h bz = load_b_frag(Wih, Kx, n0 + HID, k, lane);
    v16h bn = load_b_frag(Wih, Kx, n0 + 2 * HID, k, lane);
    i0r = wmma_f16(a0, br, i0r);
    i1r = wmma_f16(a1, br, i1r);
    i0z = wmma_f16(a0, bz, i0z);
    i1z = wmma_f16(a1, bz, i1z);
    i0n = wmma_f16(a0, bn, i0n);
    i1n = wmma_f16(a1, bn, i1n);
  }

  // gh = Hp @ Whh^T
#pragma unroll 4
  for (int k = 0; k < HID; k += 32) {
    v16h a0 = load_a_frag(Hp, HID, m0, k, lane);
    v16h a1 = load_a_frag(Hp, HID, m0 + 16, k, lane);
    v16h br = load_b_frag(Whh, HID, n0, k, lane);
    v16h bz = load_b_frag(Whh, HID, n0 + HID, k, lane);
    v16h bn = load_b_frag(Whh, HID, n0 + 2 * HID, k, lane);
    h0r = wmma_f16(a0, br, h0r);
    h1r = wmma_f16(a1, br, h1r);
    h0z = wmma_f16(a0, bz, h0z);
    h1z = wmma_f16(a1, bz, h1z);
    h0n = wmma_f16(a0, bn, h0n);
    h1n = wmma_f16(a1, bn, h1n);
  }

  // Epilogue. This lane: column n, rows mbase..mbase+7 per accumulator.
  const int n = n0 + (lane & 15);
  const int roff = (lane >> 4) << 3;  // 0 or 8
  const float bi_r = bih[n], bi_z = bih[HID + n], bi_n = bih[2 * HID + n];
  const float bh_r = bhh[n], bh_z = bhh[HID + n], bh_n = bhh[2 * HID + n];

  gru_epilogue(i0r, i0z, i0n, h0r, h0z, h0n, m0 + roff, n, bi_r, bi_z, bi_n,
               bh_r, bh_z, bh_n, nids, dyn, layer, out, Hnext);
  gru_epilogue(i1r, i1z, i1n, h1r, h1z, h1n, m0 + 16 + roff, n, bi_r, bi_z,
               bi_n, bh_r, bh_z, bh_n, nids, dyn, layer, out, Hnext);
}

// ---------------------------------------------------------------------------
// Prep kernels
// ---------------------------------------------------------------------------
__global__ void f32_to_f16_kernel(const float* __restrict__ s,
                                  _Float16* __restrict__ d, int n) {
  int i = blockIdx.x * 256 + threadIdx.x;
  if (i < n) d[i] = (_Float16)s[i];
}

// X0[i][t] = t < 512 ? conv[i][t] : static[nids[i]][t-512]   (f16, NB x 1024)
__global__ void build_x0_kernel(const float* __restrict__ conv,
                                const float* __restrict__ stat,
                                const int* __restrict__ nids,
                                _Float16* __restrict__ X0) {
  int i = blockIdx.x * 256 + threadIdx.x;
  if (i < NB * 2 * HID) {
    int row = i >> 10, t = i & 1023;
    float v = (t < HID) ? conv[(size_t)row * HID + t]
                        : stat[(size_t)nids[row] * HID + (t - HID)];
    X0[i] = (_Float16)v;
  }
}

// H[i][n] = dyn[nids[i]][layer][n]  (f16, NB x 512)
__global__ void build_h_kernel(const float* __restrict__ dyn,
                               const int* __restrict__ nids, int layer,
                               _Float16* __restrict__ H) {
  int i = blockIdx.x * 256 + threadIdx.x;
  if (i < NB * HID) {
    int row = i >> 9, n = i & 511;
    H[i] = (_Float16)dyn[(size_t)nids[row] * (2 * HID) + (size_t)layer * HID + n];
  }
}

// ---------------------------------------------------------------------------
extern "C" void kernel_launch(void* const* d_in, const int* in_sizes, int n_in,
                              void* d_out, int out_size, void* d_ws,
                              size_t ws_size, hipStream_t stream) {
  const float* conv = (const float*)d_in[0];
  const float* stat = (const float*)d_in[1];
  const float* dyn  = (const float*)d_in[2];
  const float* Wih0 = (const float*)d_in[3];
  const float* Whh0 = (const float*)d_in[4];
  const float* bih0 = (const float*)d_in[5];
  const float* bhh0 = (const float*)d_in[6];
  const float* Wih1 = (const float*)d_in[7];
  const float* Whh1 = (const float*)d_in[8];
  const float* bih1 = (const float*)d_in[9];
  const float* bhh1 = (const float*)d_in[10];
  const int*   nids = (const int*)d_in[11];
  float* out = (float*)d_out;

  // Workspace carve-out (all 256B aligned)
  char* ws = (char*)d_ws;
  size_t off = 0;
  auto carve = [&](size_t bytes) -> void* {
    void* p = ws + off;
    off += (bytes + 255) & ~(size_t)255;
    return p;
  };
  _Float16* wih0h = (_Float16*)carve((size_t)G3H * 2 * HID * 2);
  _Float16* whh0h = (_Float16*)carve((size_t)G3H * HID * 2);
  _Float16* wih1h = (_Float16*)carve((size_t)G3H * HID * 2);
  _Float16* whh1h = (_Float16*)carve((size_t)G3H * HID * 2);
  _Float16* X0    = (_Float16*)carve((size_t)NB * 2 * HID * 2);
  _Float16* H0a   = (_Float16*)carve((size_t)NB * HID * 2);  // dyn[:,0] f16
  _Float16* H0b   = (_Float16*)carve((size_t)NB * HID * 2);  // dyn[:,1] f16
  _Float16* H1    = (_Float16*)carve((size_t)NB * HID * 2);  // h_new0 f16

  // Weight conversions fp32 -> f16
  {
    int n = G3H * 2 * HID;
    f32_to_f16_kernel<<<(n + 255) / 256, 256, 0, stream>>>(Wih0, wih0h, n);
    n = G3H * HID;
    f32_to_f16_kernel<<<(n + 255) / 256, 256, 0, stream>>>(Whh0, whh0h, n);
    f32_to_f16_kernel<<<(n + 255) / 256, 256, 0, stream>>>(Wih1, wih1h, n);
    f32_to_f16_kernel<<<(n + 255) / 256, 256, 0, stream>>>(Whh1, whh1h, n);
  }
  // Activation prep
  {
    int n = NB * 2 * HID;
    build_x0_kernel<<<(n + 255) / 256, 256, 0, stream>>>(conv, stat, nids, X0);
    n = NB * HID;
    build_h_kernel<<<(n + 255) / 256, 256, 0, stream>>>(dyn, nids, 0, H0a);
    build_h_kernel<<<(n + 255) / 256, 256, 0, stream>>>(dyn, nids, 1, H0b);
  }

  // Copy the persistent table into the output, then scatter updated rows.
  hipMemcpyAsync(out, dyn, (size_t)NN * 2 * HID * sizeof(float),
                 hipMemcpyDeviceToDevice, stream);

  const int grid = (NB / 32) * 4;  // 625 row blocks x 4 col blocks = 2500
  // Layer 0: x = [conv | static], h = dyn[:,0]; writes out[nid][0] + H1 (f16)
  gru_layer_kernel<<<grid, 256, 0, stream>>>(X0, 2 * HID, H0a, wih0h, whh0h,
                                             bih0, bhh0, nids, dyn, 0, out, H1);
  // Layer 1: x = h_new0, h = dyn[:,1]; writes out[nid][1]
  gru_layer_kernel<<<grid, 256, 0, stream>>>(H1, HID, H0b, wih1h, whh1h, bih1,
                                             bhh1, nids, dyn, 1, out,
                                             (_Float16*)nullptr);
}
